// SpatialVelocityCrossAttention_25666724560957
// MI455X (gfx1250) — matrix operand order repaired and checked by reading, not compiled, not run on hardware
//
#include <hip/hip_runtime.h>
#include <hip/hip_bf16.h>
#include <math.h>

// ---------------------------------------------------------------------------
// CDNA5 (gfx1250) wave32 WMMA fragments: v_wmma_f32_16x16x32_bf16.
// A 16x32: lane m=lane&15 holds K {0..7,16..23} (lanes 0-15) / {8..15,24..31}
// B 32x16: lane n=lane&15 holds K {0..7,16..23} (lanes 0-15) / {8..15,24..31}
// C/D 16x16: acc reg r -> row r (lanes 0-15) / row 8+r (lanes 16-31), col lane&15
// ---------------------------------------------------------------------------

typedef __attribute__((ext_vector_type(16))) __bf16 v16bf;
typedef __attribute__((ext_vector_type(8)))  float  v8f;

union Frag16 {
  v16bf v;
  unsigned short u[16];
  uint4 q[2];
};
union FragC {
  v8f v;
  float f[8];
};

__device__ __forceinline__ unsigned short f2bf_raw(float f) {
  union { float f; unsigned u; } in;
  in.f = f;
  unsigned r = in.u + 0x7FFFu + ((in.u >> 16) & 1u);  // round-to-nearest-even
  return (unsigned short)(r >> 16);
}
__device__ __forceinline__ unsigned pk_bf2(float lo, float hi) {
  return (unsigned)f2bf_raw(lo) | ((unsigned)f2bf_raw(hi) << 16);
}
__device__ __forceinline__ uint4 pk_bf8(float4 a, float4 b) {
  uint4 r;
  r.x = pk_bf2(a.x, a.y);
  r.y = pk_bf2(a.z, a.w);
  r.z = pk_bf2(b.x, b.y);
  r.w = pk_bf2(b.z, b.w);
  return r;
}

// ---------------------------------------------------------------------------
// GEMM: C[M,N] = epilogue(alpha * A[M,K] . Wt[N,K]^T + bias)
// block = 256 threads (8... 4-wave M-split): block tile 64(M) x 64(N).
// Wave w computes rows [m0+16w, m0+16w+16) x 64 cols: 1 A-frag x 4 B-frags
// per 32-wide k-step -> 4 WMMAs per wave per k-step. A and B staged in LDS.
// mode 1: sigmoid f32   mode 2: bf16    mode 3: out = xsp + gamma_tiled*val
// mode 4: bf16 transposed per batch -> C[(b*N+n)*Lv + (m%Lv)]
// ---------------------------------------------------------------------------
template <bool ABF16>
__global__ __launch_bounds__(256) void wmma_gemm_kernel(
    const void* __restrict__ Aptr,
    const float* __restrict__ Wt, const float* __restrict__ bias,
    void* __restrict__ Cptr, int M, int N, int K,
    float alpha, int mode,
    const float* __restrict__ gamma, const float* __restrict__ xsp,
    int Lv, int Sv)
{
  __shared__ __align__(16) unsigned short As[64][32];
  __shared__ __align__(16) unsigned short Bs[64][32];
  const int t    = threadIdx.x;
  const int wave = t >> 5, lane = t & 31;
  const int half = lane >> 4, lm = lane & 15;
  const int m0 = blockIdx.y << 6;
  const int n0 = blockIdx.x << 6;
  const int mw = m0 + (wave << 4);
  const int kb0 = half ? 8 : 0;
  const int kb1 = half ? 24 : 16;
  const int sr = t >> 2;          // staging row 0..63
  const int scol = (t & 3) << 3;  // staging col 0,8,16,24

  FragC acc[4];
#pragma unroll
  for (int nt = 0; nt < 4; ++nt)
#pragma unroll
    for (int i = 0; i < 8; ++i) acc[nt].f[i] = 0.f;

  for (int kk = 0; kk < K; kk += 32) {
    // stage A tile 64x32 (8 bf16 per thread, one 16B store)
    if (ABF16) {
      const unsigned short* ap =
          (const unsigned short*)Aptr + (size_t)(m0 + sr) * K + kk + scol;
      *(uint4*)&As[sr][scol] = *(const uint4*)ap;
    } else {
      const float* ap = (const float*)Aptr + (size_t)(m0 + sr) * K + kk + scol;
      float4 a0 = *(const float4*)ap;
      float4 a1 = *(const float4*)(ap + 4);
      *(uint4*)&As[sr][scol] = pk_bf8(a0, a1);
    }
    // stage B tile 64(n)x32(k)
    {
      const float* bp = Wt + (size_t)(n0 + sr) * K + kk + scol;
      float4 b0 = *(const float4*)bp;
      float4 b1 = *(const float4*)(bp + 4);
      *(uint4*)&Bs[sr][scol] = pk_bf8(b0, b1);
    }
    __syncthreads();

    Frag16 fa;
    const uint4* arow = (const uint4*)&As[(wave << 4) + lm][0];
    fa.q[0] = arow[kb0 >> 3];
    fa.q[1] = arow[kb1 >> 3];

#pragma unroll
    for (int nt = 0; nt < 4; ++nt) {
      Frag16 fb;
      const uint4* brow = (const uint4*)&Bs[(nt << 4) + lm][0];
      fb.q[0] = brow[half];
      fb.q[1] = brow[2 + half];
      acc[nt].v = __builtin_amdgcn_wmma_f32_16x16x32_bf16(
          false, fa.v, false, fb.v, (short)0, acc[nt].v, false, false);
    }
    __syncthreads();
  }

#pragma unroll
  for (int nt = 0; nt < 4; ++nt) {
#pragma unroll
    for (int r = 0; r < 8; ++r) {
      int mrow = mw + r + (half << 3);
      int ncol = n0 + (nt << 4) + lm;
      float val = acc[nt].f[r] * alpha;
      if (bias) val += bias[ncol];
      size_t oidx = (size_t)mrow * N + ncol;
      if (mode == 1) {
        ((float*)Cptr)[oidx] = 1.f / (1.f + __expf(-val));
      } else if (mode == 2) {
        ((unsigned short*)Cptr)[oidx] = f2bf_raw(val);
      } else if (mode == 3) {
        int bb = mrow / Lv;
        int l  = mrow % Lv;
        size_t gidx = ((size_t)bb * Sv + (l % Sv)) * N + ncol;
        ((float*)Cptr)[oidx] = xsp[oidx] + gamma[gidx] * val;
      } else if (mode == 4) {
        int bb = mrow / Lv;
        int srow = mrow % Lv;
        ((unsigned short*)Cptr)[((size_t)bb * N + ncol) * Lv + srow] =
            f2bf_raw(val);
      } else {
        ((float*)Cptr)[oidx] = val;
      }
    }
  }
}

// ---------------------------------------------------------------------------
// Flash attention: grid (L/64, H, B), block 128 (4 waves), wave = 16 q rows.
// q pre-scaled by 1/sqrt(dk).  K chunk (32 x 64) and V chunk (64d x 32s,
// from transposed V) staged in LDS and shared by all 4 waves.
// Per chunk per wave: 4 score WMMAs + online softmax + 4 PV WMMAs.
// ---------------------------------------------------------------------------
__global__ __launch_bounds__(128) void attn_kernel(
    const unsigned short* __restrict__ qb,   // [B,L,dm] bf16
    const unsigned short* __restrict__ kbuf, // [B,S,dm] bf16
    const unsigned short* __restrict__ vT,   // [B,dm,S] bf16
    unsigned short* __restrict__ ob,         // [B,L,dm] bf16
    int B, int L, int S, int dm)
{
  __shared__ __align__(16) unsigned short Ks[32][64];      // [s][d]
  __shared__ __align__(16) unsigned short Vs[64][32];      // [d][s]
  __shared__ __align__(16) unsigned short Pbuf[4][16][32]; // per-wave P tile
  const int b = blockIdx.z, h = blockIdx.y;
  const int wave = threadIdx.x >> 5, lane = threadIdx.x & 31;
  const int half = lane >> 4, lm = lane & 15;
  const int kb0 = half ? 8 : 0;
  const int kb1 = half ? 24 : 16;
  const int l0 = (blockIdx.x << 6) + (wave << 4);
  const int hd = h << 6;
  const int t = threadIdx.x;

  // Q fragments (16x64 = two 16x32 A-frags), loaded once
  Frag16 qa0, qa1;
  {
    const unsigned short* qrow = qb + ((size_t)(b * L + l0 + lm) * dm + hd);
    qa0.q[0] = *(const uint4*)(qrow + kb0);
    qa0.q[1] = *(const uint4*)(qrow + kb1);
    qa1.q[0] = *(const uint4*)(qrow + 32 + kb0);
    qa1.q[1] = *(const uint4*)(qrow + 32 + kb1);
  }

  float mrow[8], lrow[8];
  FragC O[4];
#pragma unroll
  for (int r = 0; r < 8; ++r) { mrow[r] = -INFINITY; lrow[r] = 0.f; }
#pragma unroll
  for (int dt = 0; dt < 4; ++dt)
#pragma unroll
    for (int r = 0; r < 8; ++r) O[dt].f[r] = 0.f;

  for (int s0 = 0; s0 < S; s0 += 32) {
    // ---- stage K chunk [32s x 64d] : 16 bf16 per thread
    {
      int r = t >> 2, c = (t & 3) << 4;
      const unsigned short* kp = kbuf + ((size_t)(b * S + s0 + r) * dm + hd + c);
      *(uint4*)&Ks[r][c]     = *(const uint4*)kp;
      *(uint4*)&Ks[r][c + 8] = *(const uint4*)(kp + 8);
    }
    // ---- stage V chunk [64d x 32s] from transposed V
    {
      int r = t >> 1, c = (t & 1) << 4;
      const unsigned short* vp = vT + ((size_t)(b * dm + hd + r) * S + s0 + c);
      *(uint4*)&Vs[r][c]     = *(const uint4*)vp;
      *(uint4*)&Vs[r][c + 8] = *(const uint4*)(vp + 8);
    }
    __syncthreads();

    // ---- scores: two 16x16 tiles (s cols 0..15 / 16..31)
    FragC st[2];
#pragma unroll
    for (int tile = 0; tile < 2; ++tile) {
#pragma unroll
      for (int r = 0; r < 8; ++r) st[tile].f[r] = 0.f;
      const uint4* krow = (const uint4*)&Ks[(tile << 4) + lm][0];
      Frag16 kf0, kf1;
      kf0.q[0] = krow[half];          // d =  0..31, K {8h..8h+7, 16+8h..}
      kf0.q[1] = krow[2 + half];
      kf1.q[0] = krow[4 + half];      // d = 32..63
      kf1.q[1] = krow[6 + half];
      st[tile].v = __builtin_amdgcn_wmma_f32_16x16x32_bf16(
          false, qa0.v, false, kf0.v, (short)0, st[tile].v, false, false);
      st[tile].v = __builtin_amdgcn_wmma_f32_16x16x32_bf16(
          false, qa1.v, false, kf1.v, (short)0, st[tile].v, false, false);
    }

    // ---- online softmax: row r lives on one 16-lane half at acc index r
#pragma unroll
    for (int r = 0; r < 8; ++r) {
      float v0 = st[0].f[r], v1 = st[1].f[r];
      float cm = fmaxf(v0, v1);
#pragma unroll
      for (int off = 8; off >= 1; off >>= 1)
        cm = fmaxf(cm, __shfl_xor(cm, off, 32));
      float mnew = fmaxf(mrow[r], cm);
      float corr = __expf(mrow[r] - mnew);
      float p0 = __expf(v0 - mnew), p1 = __expf(v1 - mnew);
      float ps = p0 + p1;
#pragma unroll
      for (int off = 8; off >= 1; off >>= 1)
        ps += __shfl_xor(ps, off, 32);
      lrow[r] = lrow[r] * corr + ps;
      mrow[r] = mnew;
#pragma unroll
      for (int dt = 0; dt < 4; ++dt) O[dt].f[r] *= corr;
      int prow = r + (half << 3);
      Pbuf[wave][prow][lm]      = f2bf_raw(p0);
      Pbuf[wave][prow][lm + 16] = f2bf_raw(p1);
    }

    // ---- re-read P as A-fragment (intra-wave; DS ops are in-order)
    Frag16 pf;
    const uint4* prow4 = (const uint4*)&Pbuf[wave][lm][0];
    pf.q[0] = prow4[kb0 >> 3];
    pf.q[1] = prow4[kb1 >> 3];

    // ---- O += P . V  (V B-frags from LDS, contiguous in s)
#pragma unroll
    for (int dt = 0; dt < 4; ++dt) {
      Frag16 vf;
      const uint4* vrow = (const uint4*)&Vs[(dt << 4) + lm][0];
      vf.q[0] = vrow[half];
      vf.q[1] = vrow[2 + half];
      O[dt].v = __builtin_amdgcn_wmma_f32_16x16x32_bf16(
          false, pf.v, false, vf.v, (short)0, O[dt].v, false, false);
    }
    __syncthreads();
  }

#pragma unroll
  for (int r = 0; r < 8; ++r) {
    float inv = 1.f / lrow[r];
    int l = l0 + r + (half << 3);
#pragma unroll
    for (int dt = 0; dt < 4; ++dt)
      ob[(size_t)(b * L + l) * dm + hd + (dt << 4) + lm] =
          f2bf_raw(O[dt].f[r] * inv);
  }
}

// ---------------------------------------------------------------------------
// Three conv1ds (K=3,5,7, 64->64, 'same'): block = (b, 8 s-positions), 192 thr.
// ---------------------------------------------------------------------------
__global__ __launch_bounds__(192) void conv_ms_kernel(
    const float* __restrict__ xv,
    const float* __restrict__ w3, const float* __restrict__ b3,
    const float* __restrict__ w5, const float* __restrict__ b5,
    const float* __restrict__ w7, const float* __restrict__ b7,
    float* __restrict__ ms, int B, int S)
{
  __shared__ float xs[14][64];
  const int b = blockIdx.y;
  const int s0 = blockIdx.x << 3;
  const int t = threadIdx.x;

  for (int e = t; e < 14 * 64; e += 192) {
    int r = e >> 6, c = e & 63;
    int s = s0 - 3 + r;
    xs[r][c] = (s >= 0 && s < S) ? xv[((size_t)b * S + s) * 64 + c] : 0.f;
  }
  __syncthreads();

  const int conv = t / 64, co = t & 63;
  const float* w; const float* bb; int ksz;
  if (conv == 0)      { w = w3; bb = b3; ksz = 3; }
  else if (conv == 1) { w = w5; bb = b5; ksz = 5; }
  else                { w = w7; bb = b7; ksz = 7; }
  const int pad = ksz >> 1;

  float acc[8];
#pragma unroll
  for (int i = 0; i < 8; ++i) acc[i] = bb[co];

  for (int tap = 0; tap < ksz; ++tap) {
    int base = 3 - pad + tap;
    for (int ci = 0; ci < 64; ++ci) {
      float wv = w[((size_t)co * 64 + ci) * ksz + tap];
#pragma unroll
      for (int i = 0; i < 8; ++i) acc[i] += wv * xs[base + i][ci];
    }
  }
#pragma unroll
  for (int i = 0; i < 8; ++i)
    ms[((size_t)b * S + s0 + i) * 192 + conv * 64 + co] = acc[i];
}

// mean_rel[j,d] = (1/S) * sum_i rel[clip(j-i,-30,30)+30, d] via bucket counts
__global__ void mean_rel_kernel(const float* __restrict__ rel,
                                float* __restrict__ mr, int S, int D)
{
  int idx = blockIdx.x * blockDim.x + threadIdx.x;
  if (idx >= S * D) return;
  int j = idx / D, d = idx % D;
  float acc = 0.f;
#pragma unroll 1
  for (int rr = -30; rr <= 30; ++rr) {
    float cnt;
    if (rr == 30)        cnt = (j >= 30) ? (float)(j - 29) : 0.f;
    else if (rr == -30)  cnt = (j + 30 <= S - 1) ? (float)(S - j - 30) : 0.f;
    else                 cnt = (j - rr >= 0 && j - rr < S) ? 1.f : 0.f;
    acc += cnt * rel[(size_t)(rr + 30) * D + d];
  }
  mr[idx] = acc / (float)S;
}

// x_pe = ms + mean_rel + (diag_conv - center) / S
__global__ void relpos_kernel(const float* __restrict__ ms,
                              const float* __restrict__ mr,
                              const float* __restrict__ rel,
                              const float* __restrict__ dww,
                              const float* __restrict__ dwb,
                              float* __restrict__ xpe, int B, int S, int D)
{
  int idx = blockIdx.x * blockDim.x + threadIdx.x;
  if (idx >= B * S * D) return;
  int d = idx % D;
  int s = (idx / D) % S;
  float m = ms[idx];
  float center = m + rel[(size_t)30 * D + d];
  float left  = (s > 0)     ? ms[idx - D] + rel[(size_t)29 * D + d] : 0.f;
  float right = (s < S - 1) ? ms[idx + D] + rel[(size_t)31 * D + d] : 0.f;
  float diag = dww[d * 3 + 0] * left + dww[d * 3 + 1] * center +
               dww[d * 3 + 2] * right + dwb[d];
  xpe[idx] = m + mr[(size_t)s * D + d] + (diag - center) / (float)S;
}

// in-place LayerNorm over C=512, one block per row, 256 threads x 2 elems
__global__ __launch_bounds__(256) void ln_kernel(float* __restrict__ out,
                                                 const float* __restrict__ g,
                                                 const float* __restrict__ bta,
                                                 int C)
{
  __shared__ float red[256];
  const int row = blockIdx.x, t = threadIdx.x;
  float x0 = out[(size_t)row * C + t];
  float x1 = out[(size_t)row * C + t + 256];
  red[t] = x0 + x1;
  __syncthreads();
  for (int o = 128; o > 0; o >>= 1) {
    if (t < o) red[t] += red[t + o];
    __syncthreads();
  }
  float mean = red[0] / (float)C;
  __syncthreads();
  float d0 = x0 - mean, d1 = x1 - mean;
  red[t] = d0 * d0 + d1 * d1;
  __syncthreads();
  for (int o = 128; o > 0; o >>= 1) {
    if (t < o) red[t] += red[t + o];
    __syncthreads();
  }
  float rstd = rsqrtf(red[0] / (float)C + 1e-5f);
  out[(size_t)row * C + t]       = d0 * rstd * g[t] + bta[t];
  out[(size_t)row * C + t + 256] = d1 * rstd * g[t + 256] + bta[t + 256];
}

// ---------------------------------------------------------------------------
extern "C" void kernel_launch(void* const* d_in, const int* in_sizes, int n_in,
                              void* d_out, int out_size, void* d_ws, size_t ws_size,
                              hipStream_t stream)
{
  const float* x_spatial  = (const float*)d_in[0];
  const float* x_velocity = (const float*)d_in[1];
  const float* w_gamma    = (const float*)d_in[2];
  const float* w3 = (const float*)d_in[3];
  const float* b3 = (const float*)d_in[4];
  const float* w5 = (const float*)d_in[5];
  const float* b5 = (const float*)d_in[6];
  const float* w7 = (const float*)d_in[7];
  const float* b7 = (const float*)d_in[8];
  const float* rel_table = (const float*)d_in[9];
  const float* dw_w = (const float*)d_in[10];
  const float* dw_b = (const float*)d_in[11];
  const float* wq   = (const float*)d_in[12];
  const float* wk   = (const float*)d_in[13];
  const float* wv   = (const float*)d_in[14];
  const float* fc_w = (const float*)d_in[15];
  const float* fc_b = (const float*)d_in[16];
  const float* ln_g = (const float*)d_in[17];
  const float* ln_b = (const float*)d_in[18];

  const int B = 8, L = 1024, S = 512, sd = 512, vd = 64, dm = 512, D = 192, H = 8;

  char* ws = (char*)d_ws;
  float* gamma = (float*)(ws + 0);                          //  8 MB [B,S,dm]
  float* msbuf = (float*)(ws + 8388608);                    //  3 MB [B,S,D]
  float* mrel  = (float*)(ws + 11534336);                   // .4 MB [S,D]
  float* xpe   = (float*)(ws + 11927552);                   //  3 MB [B,S,D]
  unsigned short* qbuf = (unsigned short*)(ws + 15073280);  //  8 MB [B,L,dm]
  unsigned short* kbuf = (unsigned short*)(ws + 23461888);  //  4 MB [B,S,dm]
  unsigned short* vbuf = (unsigned short*)(ws + 27656192);  //  4 MB [B,dm,S] (transposed)
  unsigned short* abuf = (unsigned short*)(ws + 31850496);  //  8 MB [B,L,dm]
  (void)ws_size; (void)in_sizes; (void)n_in; (void)out_size;

  conv_ms_kernel<<<dim3(S / 8, B), 192, 0, stream>>>(
      x_velocity, w3, b3, w5, b5, w7, b7, msbuf, B, S);

  mean_rel_kernel<<<dim3((S * D + 255) / 256), 256, 0, stream>>>(
      rel_table, mrel, S, D);

  relpos_kernel<<<dim3((B * S * D + 255) / 256), 256, 0, stream>>>(
      msbuf, mrel, rel_table, dw_w, dw_b, xpe, B, S, D);

  // gamma = sigmoid(x_velocity . w_gamma^T)   [B*S, dm], K=64
  wmma_gemm_kernel<false><<<dim3(dm / 64, (B * S) / 64), 256, 0, stream>>>(
      x_velocity, w_gamma, nullptr, gamma, B * S, dm, vd,
      1.f, 1, nullptr, nullptr, 0, 0);

  // q = (x_spatial . wq^T) / 8 -> bf16
  wmma_gemm_kernel<false><<<dim3(dm / 64, (B * L) / 64), 256, 0, stream>>>(
      x_spatial, wq, nullptr, qbuf, B * L, dm, sd,
      0.125f, 2, nullptr, nullptr, 0, 0);

  // k = x_pe . wk^T -> bf16 [B,S,dm]
  wmma_gemm_kernel<false><<<dim3(dm / 64, (B * S) / 64), 256, 0, stream>>>(
      xpe, wk, nullptr, kbuf, B * S, dm, D, 1.f, 2, nullptr, nullptr, 0, 0);

  // v = x_pe . wv^T -> bf16 transposed [B,dm,S]  (mode 4, Lv = S)
  wmma_gemm_kernel<false><<<dim3(dm / 64, (B * S) / 64), 256, 0, stream>>>(
      xpe, wv, nullptr, vbuf, B * S, dm, D, 1.f, 4, nullptr, nullptr, S, 0);

  // flash attention -> abuf (bf16, [B,L,dm] head-interleaved)
  attn_kernel<<<dim3(L / 64, H, B), 128, 0, stream>>>(
      qbuf, kbuf, vbuf, abuf, B, L, S, dm);

  // d_out = x_spatial + gamma_tiled * (attn . fc_w^T + fc_b)
  wmma_gemm_kernel<true><<<dim3(dm / 64, (B * L) / 64), 256, 0, stream>>>(
      abuf, fc_w, fc_b, d_out, B * L, dm, dm,
      1.f, 3, gamma, x_spatial, L, S);

  // in-place LayerNorm
  ln_kernel<<<dim3(B * L), 256, 0, stream>>>((float*)d_out, ln_g, ln_b, dm);
}
